// MambaTTSDecoderLayer_61478161875535
// MI455X (gfx1250) — matrix operand order repaired and checked by reading, not compiled, not run on hardware
//
#include <hip/hip_runtime.h>
#include <hip/hip_bf16.h>
#include <math.h>

// ---------------- constants (match reference) ----------------
#define BATCH   4
#define SEQ     1024
#define TFR     512
#define DMODEL  1024
#define NHEADS  16
#define HD      64
#define DFF     4096
#define DINNER  2048
#define DSTATE  16
#define DCONV   4
#define DTRANK  64

// ---------------- WMMA types ----------------
typedef __attribute__((ext_vector_type(8)))  float          v8f;
typedef __attribute__((ext_vector_type(16))) __bf16         v16bf;
typedef __attribute__((ext_vector_type(16))) unsigned short v16s;
typedef __attribute__((ext_vector_type(8)))  unsigned short su8;

union FragU { su8 h[2]; v16s s; v16bf b; };

// ---------------- TDM support (6-arg builtin: clang-23 / therock-10.0 headers) ----------------
#if defined(__gfx1250__) && __has_builtin(__builtin_amdgcn_tensor_load_to_lds) && \
    __has_builtin(__builtin_amdgcn_s_wait_tensorcnt)
#define HAVE_TDM 1
#else
#define HAVE_TDM 0
#endif

#if HAVE_TDM
typedef unsigned int u32x4 __attribute__((ext_vector_type(4)));
typedef int          i32x8 __attribute__((ext_vector_type(8)));
typedef int          i32x4 __attribute__((ext_vector_type(4)));

// 2D tensor tile -> LDS via Tensor Data Mover.
// D# group0: count=1 | lds_addr | global_addr | type=2 ("image")
// D# group1: data_size=2B, optional LDS pad (8 dwords every 16 dwords -> row stride 96B),
//            tensor_dim0/1, tile_dim0/1, tensor_dim0_stride (elements).
// Groups 2/3 (and trailing group arg) unused for 2D tiles -> zeros.
__device__ __forceinline__ void tdm_load_2d(const void* gptr, unsigned lds_off,
                                            unsigned tile_d0, unsigned tile_d1,
                                            unsigned stride_elems, bool pad)
{
    unsigned long long ga = (unsigned long long)(size_t)gptr;
    u32x4 g0;
    g0[0] = 1u;                                            // count=1 (valid user descriptor)
    g0[1] = lds_off;                                       // lds_addr (bytes)
    g0[2] = (unsigned)(ga & 0xFFFFFFFFu);                  // global_addr[31:0]
    g0[3] = (unsigned)((ga >> 32) & 0x01FFFFFFu)           // global_addr[56:32]
          | (2u << 30);                                    // type=2
    unsigned w0 = (1u << 16);                              // data_size=1 -> 2 bytes
    if (pad) w0 |= (1u << 20)                              // pad_enable
               |  (3u << 22)                               // pad_interval: 16 DWORDs
               |  (7u << 25);                              // pad_amount:   8 DWORDs
    i32x8 g1;
    g1[0] = (int)w0;                                       // workgroup_mask=0 | data_size | pad
    g1[1] = (int)((tile_d0 & 0xFFFFu) << 16);              // tensor_dim0[15:0] (== tile_d0)
    g1[2] = (int)(((tile_d0 >> 16) & 0xFFFFu)              // tensor_dim0[31:16]
          |       ((tile_d1 & 0xFFFFu) << 16));            // tensor_dim1[15:0]
    g1[3] = (int)(((tile_d1 >> 16) & 0xFFFFu)              // tensor_dim1[31:16]
          |       ((tile_d0 & 0xFFFFu) << 16));            // tile_dim0
    g1[4] = (int)(tile_d1 & 0xFFFFu);                      // tile_dim1 | tile_dim2=0
    g1[5] = (int)stride_elems;                             // tensor_dim0_stride[31:0]
    g1[6] = 0;                                             // stride hi | dim1_stride lo
    g1[7] = 0;
    const i32x4 z4  = {0, 0, 0, 0};
    const i32x8 z8v = {0, 0, 0, 0, 0, 0, 0, 0};
    __builtin_amdgcn_tensor_load_to_lds(g0, g1, z4, z4, z8v, 0);
}
__device__ __forceinline__ unsigned lds_offset_of(const void* p) {
    return (unsigned)(size_t)p;    // generic LDS pointer: low 32 bits = LDS byte offset
}
#endif

__device__ __forceinline__ unsigned short f32_to_bf16(float f) {
    union { float f; unsigned u; } v; v.f = f;
    unsigned r = v.u + 0x7FFFu + ((v.u >> 16) & 1u);   // round-to-nearest-even
    return (unsigned short)(r >> 16);
}
__device__ __forceinline__ float bf16_to_f32(unsigned short h) {
    union { unsigned u; float f; } v; v.u = ((unsigned)h) << 16;
    return v.f;
}

__device__ __forceinline__ v8f wmma_bf16(v16bf a, v16bf b, v8f c) {
    // D = A(16x32 bf16) * B(32x16 bf16) + C(16x16 f32)
    return __builtin_amdgcn_wmma_f32_16x16x32_bf16(false, a, false, b,
                                                   (short)0, c, false, false);
}

// A fragment (16x32, bf16): row = lane&15; lanes>=16 take K offset +8.
__device__ __forceinline__ v16bf load_a16x32(const unsigned short* base, int rowStride, int lane) {
    const unsigned short* rp = base + (size_t)(lane & 15) * rowStride + ((lane & 16) ? 8 : 0);
    FragU u;
    u.h[0] = *(const su8*)rp;
    u.h[1] = *(const su8*)(rp + 16);
    return u.b;
}
// B fragment (32x16, bf16), stored as B^T[n][k]; col = lane&15, lanes>=16 take K offset +8.
__device__ __forceinline__ v16bf load_b32x16(const unsigned short* base, int colStride, int lane) {
    const unsigned short* rp = base + (size_t)(lane & 15) * colStride + ((lane & 16) ? 8 : 0);
    FragU u;
    u.h[0] = *(const su8*)rp;
    u.h[1] = *(const su8*)(rp + 16);
    return u.b;
}

// ---------------- generic WMMA GEMM ----------------
// C(MxN,f32) = act(A(MxK,bf16) @ B + bias) + residual ; optional bf16 copy out.
// transB==0: B is KxN row-major.  transB==1: B given as NxK row-major (i.e. use B^T).
#define LSTR 48   // LDS row stride (bf16 elems): 96B, 16B-aligned; == TDM pad layout

__global__ __launch_bounds__(128) void gemm_wmma(
    const unsigned short* __restrict__ A, const unsigned short* __restrict__ B,
    int M, int N, int K, int transB,
    const float* __restrict__ bias, int act,
    const float* __restrict__ residual,
    float* __restrict__ outF, unsigned short* __restrict__ outB)
{
    __shared__ __align__(16) unsigned short As[64 * LSTR];
    __shared__ __align__(16) unsigned short Bs[64 * LSTR];

    const int tid  = threadIdx.x;
    const int lane = tid & 31;
    const int wave = tid >> 5;
    const int waveM = wave >> 1, waveN = wave & 1;
    const int m0 = blockIdx.y * 64, n0 = blockIdx.x * 64;

    const v8f vzero = {0.f,0.f,0.f,0.f,0.f,0.f,0.f,0.f};
    v8f a00 = vzero, a01 = vzero, a10 = vzero, a11 = vzero;
    const su8 z8 = {0,0,0,0,0,0,0,0};

    for (int k0 = 0; k0 < K; k0 += 32) {
        __syncthreads();   // previous iteration's LDS reads complete before overwrite
#if HAVE_TDM
        // TDM path: wave 0 issues 2D tile DMA(s); LDS pad reproduces the 48-elem row stride.
        if (wave == 0) {
            tdm_load_2d(A + (size_t)m0 * K + k0, lds_offset_of(As), 32u, 64u, (unsigned)K, true);
            if (transB)
                tdm_load_2d(B + (size_t)n0 * K + k0, lds_offset_of(Bs), 32u, 64u, (unsigned)K, true);
            __builtin_amdgcn_s_wait_tensorcnt(0);
        }
#else
        {   // A tile 64x32 -> As (row-major)
            int r  = tid >> 1;
            int cb = (tid & 1) * 16;
            su8 x0 = z8, x1 = z8;
            if (m0 + r < M) {
                const unsigned short* p = A + (size_t)(m0 + r) * K + k0 + cb;
                x0 = *(const su8*)p;
                x1 = *(const su8*)(p + 8);
                if (k0 + 32 < K) __builtin_prefetch(p + 32, 0, 1);  // global_prefetch_b8
            }
            *(su8*)&As[r * LSTR + cb]     = x0;
            *(su8*)&As[r * LSTR + cb + 8] = x1;
        }
        if (transB) {   // B stored NxK: rows are contiguous in K
            int n  = tid >> 1;
            int kb = (tid & 1) * 16;
            su8 x0 = z8, x1 = z8;
            if (n0 + n < N) {
                const unsigned short* p = B + (size_t)(n0 + n) * K + k0 + kb;
                x0 = *(const su8*)p;
                x1 = *(const su8*)(p + 8);
                if (k0 + 32 < K) __builtin_prefetch(p + 32, 0, 1);
            }
            *(su8*)&Bs[n * LSTR + kb]     = x0;
            *(su8*)&Bs[n * LSTR + kb + 8] = x1;
        }
#endif
        if (!transB) {  // B stored KxN: transpose into Bs[n][k] (TDM cannot transpose)
            int n  = tid & 63;
            int kb = (tid >> 6) * 16;
            #pragma unroll
            for (int j = 0; j < 16; ++j) {
                unsigned short v = 0;
                if (n0 + n < N) v = B[(size_t)(k0 + kb + j) * N + n0 + n];
                Bs[n * LSTR + kb + j] = v;
            }
        }
        __syncthreads();

        v16bf fa0 = load_a16x32(As + (waveM * 32 +  0) * LSTR, LSTR, lane);
        v16bf fa1 = load_a16x32(As + (waveM * 32 + 16) * LSTR, LSTR, lane);
        v16bf fb0 = load_b32x16(Bs + (waveN * 32 +  0) * LSTR, LSTR, lane);
        v16bf fb1 = load_b32x16(Bs + (waveN * 32 + 16) * LSTR, LSTR, lane);

        a00 = wmma_bf16(fa0, fb0, a00);
        a01 = wmma_bf16(fa0, fb1, a01);
        a10 = wmma_bf16(fa1, fb0, a10);
        a11 = wmma_bf16(fa1, fb1, a11);
    }

    // epilogue: C layout -> M = vgpr + 8*(lane>=16), N = lane&15
    const int mlane = (lane & 16) ? 8 : 0;
    const int nlane = lane & 15;
    v8f accs[2][2] = { { a00, a01 }, { a10, a11 } };
    #pragma unroll
    for (int tm = 0; tm < 2; ++tm)
    #pragma unroll
    for (int tn = 0; tn < 2; ++tn) {
        #pragma unroll
        for (int r = 0; r < 8; ++r) {
            int m = m0 + waveM * 32 + tm * 16 + mlane + r;
            int n = n0 + waveN * 32 + tn * 16 + nlane;
            if (m < M && n < N) {
                float v = accs[tm][tn][r];
                if (bias) v += bias[n];
                if (act == 1)      v = 0.5f * v * (1.0f + erff(v * 0.70710678f)); // exact GELU
                else if (act == 2) v = v / (1.0f + __expf(-v));                   // SiLU
                else if (act == 3) v = (v > 20.f) ? v : log1pf(__expf(v));        // softplus
                if (residual) v += residual[(size_t)m * N + n];
                if (outF) outF[(size_t)m * N + n] = v;
                if (outB) outB[(size_t)m * N + n] = f32_to_bf16(v);
            }
        }
    }
}

// ---------------- LayerNorm (rows of 1024) -> bf16 ----------------
__global__ __launch_bounds__(128) void layernorm_bf16(
    const float* __restrict__ x, const float* __restrict__ g, const float* __restrict__ b,
    unsigned short* __restrict__ out, int D)
{
    int row = blockIdx.x, tid = threadIdx.x;
    const float* xr = x + (size_t)row * D;
    float s = 0.f, s2 = 0.f;
    for (int i = tid; i < D; i += 128) { float v = xr[i]; s += v; s2 += v * v; }
    __shared__ float r1[128], r2[128];
    r1[tid] = s; r2[tid] = s2; __syncthreads();
    for (int off = 64; off > 0; off >>= 1) {
        if (tid < off) { r1[tid] += r1[tid + off]; r2[tid] += r2[tid + off]; }
        __syncthreads();
    }
    float mean = r1[0] / D;
    float var  = r2[0] / D - mean * mean;
    float inv  = rsqrtf(var + 1e-5f);
    for (int i = tid; i < D; i += 128) {
        float v = (xr[i] - mean) * inv * g[i] + b[i];
        out[(size_t)row * D + i] = f32_to_bf16(v);
    }
}

// ---------------- f32 -> bf16 converts ----------------
__global__ void to_bf16(const float* __restrict__ s, unsigned short* __restrict__ d, long long n) {
    long long i = (long long)blockIdx.x * 256 + threadIdx.x;
    if (i < n) d[i] = f32_to_bf16(s[i]);
}
__global__ void slice_bf16(const float* __restrict__ src, unsigned short* __restrict__ dst,
                           int rows, int width, int stride, int off) {
    long long i = (long long)blockIdx.x * 256 + threadIdx.x;
    if (i >= (long long)rows * width) return;
    int r = (int)(i / width), c = (int)(i % width);
    dst[i] = f32_to_bf16(src[(size_t)r * stride + off + c]);
}

// ---------------- causal depthwise conv (width 4) + SiLU ----------------
__global__ void conv_silu(const unsigned short* __restrict__ xz, const float* __restrict__ w,
                          const float* __restrict__ bias,
                          float* __restrict__ xcf, unsigned short* __restrict__ xcb)
{
    long long idx = (long long)blockIdx.x * 256 + threadIdx.x;   // over B*SEQ*DINNER
    if (idx >= (long long)BATCH * SEQ * DINNER) return;
    int c = (int)(idx & (DINNER - 1));
    long long bt = idx >> 11;
    int t = (int)(bt & (SEQ - 1));
    long long bb = bt >> 10;
    float acc = bias[c];
    #pragma unroll
    for (int j = 0; j < DCONV; ++j) {
        int tt = t - (DCONV - 1) + j;
        if (tt >= 0)
            acc += w[c * DCONV + j] * bf16_to_f32(xz[((bb * SEQ + tt) * (2 * DINNER)) + c]);
    }
    float s = acc / (1.f + __expf(-acc));
    xcf[idx] = s;
    xcb[idx] = f32_to_bf16(s);
}

// ---------------- selective scan (parallel over B x DINNER, sequential T) ----------------
__global__ __launch_bounds__(256) void mamba_scan(
    const float* __restrict__ xdbl, const float* __restrict__ dt,
    const float* __restrict__ xc, const unsigned short* __restrict__ xz,
    const float* __restrict__ A_log, const float* __restrict__ Dp,
    unsigned short* __restrict__ ybf, float* __restrict__ stateOut)
{
    int c = blockIdx.x * 256 + threadIdx.x;   // channel
    int b = blockIdx.y;
    float Ac[DSTATE], st[DSTATE];
    #pragma unroll
    for (int s = 0; s < DSTATE; ++s) { Ac[s] = -__expf(A_log[c * DSTATE + s]); st[s] = 0.f; }
    float Dc = Dp[c];
    __shared__ float Bsh[DSTATE], Csh[DSTATE];
    for (int t = 0; t < SEQ; ++t) {
        __syncthreads();
        if (threadIdx.x < 2 * DSTATE) {
            float v = xdbl[((size_t)(b * SEQ + t)) * (DTRANK + 2 * DSTATE) + DTRANK + threadIdx.x];
            if (threadIdx.x < DSTATE) Bsh[threadIdx.x] = v; else Csh[threadIdx.x - DSTATE] = v;
        }
        __syncthreads();
        size_t base = (size_t)(b * SEQ + t);
        float dtv = dt[base * DINNER + c];
        float u   = xc[base * DINNER + c];
        float y = 0.f;
        #pragma unroll
        for (int s = 0; s < DSTATE; ++s) {
            float dA = __expf(dtv * Ac[s]);
            st[s] = dA * st[s] + (dtv * Bsh[s]) * u;
            y += st[s] * Csh[s];
        }
        y += u * Dc;
        float z = bf16_to_f32(xz[base * (2 * DINNER) + DINNER + c]);
        y *= z / (1.f + __expf(-z));
        ybf[base * DINNER + c] = f32_to_bf16(y);
    }
    #pragma unroll
    for (int s = 0; s < DSTATE; ++s)
        stateOut[((size_t)(b * DINNER + c)) * DSTATE + s] = st[s];
}

// ---------------- fused masked cross-attention ----------------
// grid = (SEQ/16, NHEADS, BATCH), block = 128 (4 waves).
__global__ __launch_bounds__(128) void attn_fused(
    const unsigned short* __restrict__ qm, const unsigned short* __restrict__ km,
    const unsigned short* __restrict__ vm, const unsigned char* __restrict__ mask,
    unsigned short* __restrict__ om)
{
    __shared__ __align__(16) float          sc[16 * TFR];
    __shared__ __align__(16) unsigned short pb[16 * TFR];
    __shared__ __align__(16) unsigned short vt[HD * LSTR];
    __shared__ float red[16 * 8];

    const int b = blockIdx.z, h = blockIdx.y, q0 = blockIdx.x * 16;
    const int tid = threadIdx.x, lane = tid & 31, wave = tid >> 5;
    const v8f vzero = {0.f,0.f,0.f,0.f,0.f,0.f,0.f,0.f};

    // phase 1: scores = (Q @ K^T) * 1/sqrt(HD), masked
    for (int kt = wave; kt < TFR / 16; kt += 4) {
        int t0 = kt * 16;
        v8f acc = vzero;
        #pragma unroll
        for (int ks = 0; ks < HD; ks += 32) {
            v16bf fa = load_a16x32(qm + ((size_t)(b * SEQ + q0) * DMODEL + h * HD + ks), DMODEL, lane);
            v16bf fb = load_b32x16(km + ((size_t)(b * TFR + t0) * DMODEL + h * HD + ks), DMODEL, lane);
            acc = wmma_bf16(fa, fb, acc);
        }
        int mrow = (lane & 16) ? 8 : 0;
        int tok  = t0 + (lane & 15);
        float mv = mask[b * TFR + tok] ? 0.f : -1e30f;
        #pragma unroll
        for (int r = 0; r < 8; ++r)
            sc[(mrow + r) * TFR + tok] = acc[r] * 0.125f + mv;
    }
    __syncthreads();

    // phase 2: row softmax over TF, write bf16 probs
    {
        int row = tid & 15, chunk = tid >> 4;          // 8 chunks x 64 cols
        const float* srow = sc + row * TFR + chunk * 64;
        float mx = -1e30f;
        for (int c = 0; c < 64; ++c) mx = fmaxf(mx, srow[c]);
        red[row * 8 + chunk] = mx; __syncthreads();
        float rmx = -1e30f;
        #pragma unroll
        for (int j = 0; j < 8; ++j) rmx = fmaxf(rmx, red[row * 8 + j]);
        __syncthreads();
        float sum = 0.f;
        for (int c = 0; c < 64; ++c) sum += __expf(srow[c] - rmx);
        red[row * 8 + chunk] = sum; __syncthreads();
        float rs = 0.f;
        #pragma unroll
        for (int j = 0; j < 8; ++j) rs += red[row * 8 + j];
        float inv = 1.0f / rs;
        for (int c = 0; c < 64; ++c)
            pb[row * TFR + chunk * 64 + c] = f32_to_bf16(__expf(srow[c] - rmx) * inv);
    }

    // phase 3: O = P @ V, V tile transposed through LDS
    v8f oacc = vzero;
    for (int ks = 0; ks < TFR; ks += 32) {
        __syncthreads();
        {
            int kl = tid & 31, db = (tid >> 5) * 16;
            const unsigned short* vp = vm + ((size_t)(b * TFR + ks + kl) * DMODEL + h * HD + db);
            #pragma unroll
            for (int j = 0; j < 16; ++j) vt[(db + j) * LSTR + kl] = vp[j];
        }
        __syncthreads();
        v16bf fa = load_a16x32(pb + ks, TFR, lane);
        v16bf fb = load_b32x16(vt + (wave * 16) * LSTR, LSTR, lane);
        oacc = wmma_bf16(fa, fb, oacc);
    }
    int mrow = (lane & 16) ? 8 : 0;
    int n    = wave * 16 + (lane & 15);
    #pragma unroll
    for (int r = 0; r < 8; ++r)
        om[((size_t)(b * SEQ + q0 + mrow + r)) * DMODEL + h * HD + n] = f32_to_bf16(oacc[r]);
    (void)mrow;
}

// ---------------- host orchestration ----------------
extern "C" void kernel_launch(void* const* d_in, const int* in_sizes, int n_in,
                              void* d_out, int out_size, void* d_ws, size_t ws_size,
                              hipStream_t stream) {
    const float* x        = (const float*)d_in[0];
    const float* styled   = (const float*)d_in[1];
    const unsigned char* smask = (const unsigned char*)d_in[2];
    const float* ln1g = (const float*)d_in[3],  *ln1b = (const float*)d_in[4];
    const float* ln2g = (const float*)d_in[5],  *ln2b = (const float*)d_in[6];
    const float* ln3g = (const float*)d_in[7],  *ln3b = (const float*)d_in[8];
    const float* w_in   = (const float*)d_in[9];
    const float* convw  = (const float*)d_in[10], *convb = (const float*)d_in[11];
    const float* w_xprj = (const float*)d_in[12];
    const float* w_dt   = (const float*)d_in[13], *dt_b  = (const float*)d_in[14];
    const float* A_log  = (const float*)d_in[15], *m_D   = (const float*)d_in[16];
    const float* w_out  = (const float*)d_in[17];
    const float* w_ain  = (const float*)d_in[18], *b_ain = (const float*)d_in[19];
    const float* w_aout = (const float*)d_in[20], *b_aout= (const float*)d_in[21];
    const float* w_ff1  = (const float*)d_in[22], *b_ff1 = (const float*)d_in[23];
    const float* w_ff2  = (const float*)d_in[24], *b_ff2 = (const float*)d_in[25];

    const int ROWS = BATCH * SEQ;         // 4096
    const int KVROWS = BATCH * TFR;       // 2048

    char* p = (char*)d_ws;
    auto alloc = [&](size_t bytes) -> void* {
        void* r = (void*)p; p += (bytes + 255) & ~((size_t)255); return r;
    };
    auto allocU = [&](size_t n) { return (unsigned short*)alloc(n * 2); };
    auto allocF = [&](size_t n) { return (float*)alloc(n * 4); };

    // bf16 weight pool
    unsigned short* wb_in   = allocU((size_t)DMODEL * 2 * DINNER);
    unsigned short* wb_xprj = allocU((size_t)DINNER * (DTRANK + 2 * DSTATE));
    unsigned short* wb_dt   = allocU((size_t)DTRANK * DINNER);
    unsigned short* wb_out  = allocU((size_t)DINNER * DMODEL);
    unsigned short* wb_ain  = allocU((size_t)3 * DMODEL * DMODEL);
    unsigned short* wb_aout = allocU((size_t)DMODEL * DMODEL);
    unsigned short* wb_ff1  = allocU((size_t)DMODEL * DFF);
    unsigned short* wb_ff2  = allocU((size_t)DFF * DMODEL);
    // activations
    unsigned short* h_bf    = allocU((size_t)ROWS * DMODEL);
    unsigned short* styl_bf = allocU((size_t)KVROWS * DMODEL);
    unsigned short* xz_bf   = allocU((size_t)ROWS * 2 * DINNER);
    float*          xc_f    = allocF((size_t)ROWS * DINNER);
    unsigned short* xc_bf   = allocU((size_t)ROWS * DINNER);
    float*          xdbl_f  = allocF((size_t)ROWS * (DTRANK + 2 * DSTATE));
    unsigned short* dtin_bf = allocU((size_t)ROWS * DTRANK);
    float*          dt_f    = allocF((size_t)ROWS * DINNER);
    unsigned short* y_bf    = allocU((size_t)ROWS * DINNER);
    unsigned short* q_bf    = allocU((size_t)ROWS * DMODEL);
    unsigned short* k_bf    = allocU((size_t)KVROWS * DMODEL);
    unsigned short* v_bf    = allocU((size_t)KVROWS * DMODEL);
    unsigned short* o_bf    = allocU((size_t)ROWS * DMODEL);
    float*          x1_f    = allocF((size_t)ROWS * DMODEL);
    float*          x2_f    = allocF((size_t)ROWS * DMODEL);
    unsigned short* ffh_bf  = allocU((size_t)ROWS * DFF);

    float* out_x     = (float*)d_out;
    float* out_state = out_x + (size_t)ROWS * DMODEL;

    auto cvt = [&](const float* s, unsigned short* d, long long n) {
        to_bf16<<<(unsigned)((n + 255) / 256), 256, 0, stream>>>(s, d, n);
    };
    auto gemm = [&](const unsigned short* A, const unsigned short* Bm, int M, int N, int K,
                    int tB, const float* bias, int act, const float* res,
                    float* oF, unsigned short* oB) {
        dim3 grid((N + 63) / 64, (M + 63) / 64);
        gemm_wmma<<<grid, 128, 0, stream>>>(A, Bm, M, N, K, tB, bias, act, res, oF, oB);
    };

    // weight + input converts
    cvt(w_in,   wb_in,   (long long)DMODEL * 2 * DINNER);
    cvt(w_xprj, wb_xprj, (long long)DINNER * (DTRANK + 2 * DSTATE));
    cvt(w_dt,   wb_dt,   (long long)DTRANK * DINNER);
    cvt(w_out,  wb_out,  (long long)DINNER * DMODEL);
    cvt(w_ain,  wb_ain,  (long long)3 * DMODEL * DMODEL);
    cvt(w_aout, wb_aout, (long long)DMODEL * DMODEL);
    cvt(w_ff1,  wb_ff1,  (long long)DMODEL * DFF);
    cvt(w_ff2,  wb_ff2,  (long long)DFF * DMODEL);
    cvt(styled, styl_bf, (long long)KVROWS * DMODEL);

    // ---- Mamba branch ----
    layernorm_bf16<<<ROWS, 128, 0, stream>>>(x, ln1g, ln1b, h_bf, DMODEL);
    gemm(h_bf, wb_in, ROWS, 2 * DINNER, DMODEL, 0, nullptr, 0, nullptr, nullptr, xz_bf);
    {
        long long n = (long long)ROWS * DINNER;
        conv_silu<<<(unsigned)((n + 255) / 256), 256, 0, stream>>>(xz_bf, convw, convb, xc_f, xc_bf);
    }
    gemm(xc_bf, wb_xprj, ROWS, DTRANK + 2 * DSTATE, DINNER, 0, nullptr, 0, nullptr, xdbl_f, nullptr);
    {
        long long n = (long long)ROWS * DTRANK;
        slice_bf16<<<(unsigned)((n + 255) / 256), 256, 0, stream>>>(
            xdbl_f, dtin_bf, ROWS, DTRANK, DTRANK + 2 * DSTATE, 0);
    }
    gemm(dtin_bf, wb_dt, ROWS, DINNER, DTRANK, 0, dt_b, /*softplus*/3, nullptr, dt_f, nullptr);
    mamba_scan<<<dim3(DINNER / 256, BATCH), 256, 0, stream>>>(
        xdbl_f, dt_f, xc_f, xz_bf, A_log, m_D, y_bf, out_state);
    gemm(y_bf, wb_out, ROWS, DMODEL, DINNER, 0, nullptr, 0, /*res*/x, x1_f, nullptr);

    // ---- cross-attention branch ----
    layernorm_bf16<<<ROWS, 128, 0, stream>>>(x1_f, ln2g, ln2b, h_bf, DMODEL);
    gemm(h_bf,    wb_ain,                           ROWS,   DMODEL, DMODEL, 1, b_ain,              0, nullptr, nullptr, q_bf);
    gemm(styl_bf, wb_ain + (size_t)DMODEL * DMODEL, KVROWS, DMODEL, DMODEL, 1, b_ain + DMODEL,     0, nullptr, nullptr, k_bf);
    gemm(styl_bf, wb_ain + (size_t)2*DMODEL*DMODEL, KVROWS, DMODEL, DMODEL, 1, b_ain + 2 * DMODEL, 0, nullptr, nullptr, v_bf);
    attn_fused<<<dim3(SEQ / 16, NHEADS, BATCH), 128, 0, stream>>>(q_bf, k_bf, v_bf, smask, o_bf);
    gemm(o_bf, wb_aout, ROWS, DMODEL, DMODEL, 1, b_aout, 0, /*res*/x1_f, x2_f, nullptr);

    // ---- FFN branch ----
    layernorm_bf16<<<ROWS, 128, 0, stream>>>(x2_f, ln3g, ln3b, h_bf, DMODEL);
    gemm(h_bf,   wb_ff1, ROWS, DFF,    DMODEL, 0, b_ff1, /*gelu*/1, nullptr, nullptr, ffh_bf);
    gemm(ffh_bf, wb_ff2, ROWS, DMODEL, DFF,    0, b_ff2, 0, /*res*/x2_f, out_x, nullptr);

    (void)in_sizes; (void)n_in; (void)out_size; (void)ws_size;
}
